// RGTLayer_67310727463213
// MI455X (gfx1250) — matrix operand order repaired and checked by reading, not compiled, not run on hardware
//
#include <hip/hip_runtime.h>
#include <hip/hip_bf16.h>

// ---- Problem constants (match reference) ------------------------------------
#define B_    16
#define NE_   512
#define NIN_  1024
#define NOUT_ 1024
#define H_    16
#define HD_   64
#define M_    (B_ * NE_)                    // 8192 rows
#define EOFF_ ((size_t)B_ * NE_ * NOUT_)    // offset of `e` in d_out (floats)

typedef _Float16 v16h __attribute__((ext_vector_type(16)));
typedef _Float16 v8h  __attribute__((ext_vector_type(8)));
typedef float    v8f  __attribute__((ext_vector_type(8)));
typedef int      v4i  __attribute__((vector_size(4 * sizeof(int))));

// ---- CDNA5 async global->LDS copy (ASYNCcnt-tracked) -------------------------
#if __has_builtin(__builtin_amdgcn_global_load_async_to_lds_b128) && \
    __has_builtin(__builtin_amdgcn_s_wait_asynccnt)
#define HAS_ASYNC 1
#else
#define HAS_ASYNC 0
#endif

#if HAS_ASYNC
#define WAIT_ASYNC(n) __builtin_amdgcn_s_wait_asynccnt(n)
#else
#define WAIT_ASYNC(n) ((void)0)
#endif

// Copy 16B per lane from global to LDS. One wave instruction per call (async path).
__device__ __forceinline__ void copy16_g2l(const _Float16* g, _Float16* l) {
#if HAS_ASYNC
    __builtin_amdgcn_global_load_async_to_lds_b128(
        (__attribute__((address_space(1))) v4i*)(g),
        (__attribute__((address_space(3))) v4i*)(l),
        0 /*offset*/, 0 /*cpol*/);
#else
    *(v8h*)l = *(const v8h*)g;
#endif
}

// ---- WMMA fragment loaders (CDNA5 16x16x32 f16 layouts, wave32) -------------
// A: 16x32 (MxK), row-major [M, lda] (K contiguous) -> two 16B loads per lane.
__device__ __forceinline__ v16h load_a_frag(const _Float16* __restrict__ A, int lda,
                                            int m0, int k0, int lane) {
    int m  = m0 + (lane & 15);
    int kb = k0 + ((lane >> 4) << 4);
    const _Float16* p = A + (size_t)m * lda + kb;
    union { v16h v; v8h h[2]; } u;
    u.h[0] = *(const v8h*)(p);
    u.h[1] = *(const v8h*)(p + 8);
    return u.v;
}

// B: 32x16 (KxN) from a [N, ldb] (K-contiguous, "NT") matrix -> one 32B load/lane.
__device__ __forceinline__ v16h load_b_frag(const _Float16* __restrict__ Bw, int ldb,
                                            int n0, int k0, int lane) {
    int n  = n0 + (lane & 15);
    int kb = k0 + ((lane >> 4) << 4);
    return *(const v16h*)(Bw + (size_t)n * ldb + kb);
}

// Same B fragment, but from an LDS-staged [64][32] half tile.
__device__ __forceinline__ v16h load_b_frag_lds(const _Float16* buf, int nbase, int lane) {
    int n  = nbase + (lane & 15);
    int kb = (lane >> 4) << 4;
    return *(const v16h*)(buf + n * 32 + kb);
}

__device__ __forceinline__ v8f wmma_f16(v16h a, v16h b, v8f c) {
    return __builtin_amdgcn_wmma_f32_16x16x32_f16(false, a, false, b, (short)0, c, false, false);
}

// ---- LayerNorm (ddof=1) ------------------------------------------------------
template <typename OutT>
__global__ __launch_bounds__(256) void norm_kernel(const float* __restrict__ X,
                                                   const float* __restrict__ alpha,
                                                   const float* __restrict__ beta,
                                                   OutT* __restrict__ out, int N) {
    int row = blockIdx.x;
    const float* x = X + (size_t)row * N;
    int t = threadIdx.x, lane = t & 31, wave = t >> 5;
    float vals[4], s = 0.f, ss = 0.f;
    #pragma unroll
    for (int i = 0; i < 4; ++i) {
        float v = x[t + i * 256];
        vals[i] = v; s += v; ss += v * v;
    }
    __shared__ float rs[8], rss[8];
    #pragma unroll
    for (int off = 16; off > 0; off >>= 1) {
        s  += __shfl_xor(s, off, 32);
        ss += __shfl_xor(ss, off, 32);
    }
    if (lane == 0) { rs[wave] = s; rss[wave] = ss; }
    __syncthreads();
    if (wave == 0) {
        s  = (lane < 8) ? rs[lane]  : 0.f;
        ss = (lane < 8) ? rss[lane] : 0.f;
        #pragma unroll
        for (int off = 4; off > 0; off >>= 1) {
            s  += __shfl_xor(s, off, 32);
            ss += __shfl_xor(ss, off, 32);
        }
        if (lane == 0) { rs[0] = s; rss[0] = ss; }
    }
    __syncthreads();
    float mean = rs[0] / (float)N;
    float var  = (rss[0] - rs[0] * mean) / (float)(N - 1);   // unbiased (ddof=1)
    float inv  = 1.f / (sqrtf(fmaxf(var, 0.f)) + 1e-6f);
    #pragma unroll
    for (int i = 0; i < 4; ++i) {
        int c = t + i * 256;
        out[(size_t)row * N + c] = (OutT)(alpha[c] * (vals[i] - mean) * inv + beta[c]);
    }
}

// ---- f32 -> f16 weight convert ----------------------------------------------
__global__ __launch_bounds__(256) void cvt_f32_f16(const float* __restrict__ in,
                                                   _Float16* __restrict__ out, int n) {
    int i = blockIdx.x * 256 + threadIdx.x;
    if (i < n) out[i] = (_Float16)in[i];
}

// ---- Shared store helper for the big GEMMs ----------------------------------
// mode 0: Q/K scatter f16 [B,H,NE,HD]; mode 1: V^T scatter f16 [B,H,HD,NE];
// mode 2: f32 row-major [M, NOUT] (+bias) for the WO projection.
__device__ __forceinline__ void store_proj_tile(v8f acc, int m0, int n0, int lane,
                                                const float* __restrict__ bias,
                                                _Float16* __restrict__ out16,
                                                float* __restrict__ out32, int mode) {
    int n  = n0 + (lane & 15);
    int mb = m0 + ((lane >> 4) << 3);
    float bv = bias[n];
    int h = n >> 6, d = n & 63;
    #pragma unroll
    for (int r = 0; r < 8; ++r) {
        int m = mb + r;
        float v = acc[r] + bv;
        if (mode == 0) {
            int b = m >> 9, q = m & 511;
            out16[(((size_t)(b * H_ + h) * NE_) + q) * HD_ + d] = (_Float16)v;
        } else if (mode == 1) {
            int b = m >> 9, q = m & 511;
            out16[(((size_t)(b * H_ + h) * HD_) + d) * NE_ + q] = (_Float16)v;
        } else {
            out32[(size_t)m * NOUT_ + n] = v;
        }
    }
}

// ---- Big GEMM: C = A[8192,1024] @ W[1024,1024]^T (+bias), blocked -----------
// Block = 8 waves (4M x 2N), block tile 128M x 64N, wave tile 32x32 (4 WMMA/step).
// Weight tile (64N x 32K) is double-buffered in LDS via async global->LDS copies.
__global__ __launch_bounds__(256) void gemm_big_kernel(const _Float16* __restrict__ A,
                                                       const _Float16* __restrict__ W,
                                                       const float* __restrict__ bias,
                                                       _Float16* __restrict__ out16,
                                                       float* __restrict__ out32, int mode) {
    __shared__ _Float16 ldsB[2][64 * 32];   // 8 KB double buffer
    int tid = threadIdx.x;
    int lane = tid & 31, w = tid >> 5;
    int wm = w >> 1, wn = w & 1;
    int bx = blockIdx.x & 15, by = blockIdx.x >> 4;      // 16 N-blocks, 64 M-blocks
    int blockN0 = bx << 6;
    int m0 = (by << 7) + (wm << 5);
    int nL = wn << 5;                                    // wave N offset within block
    int n0 = blockN0 + nL;

    // per-thread cooperative copy coords: 256 threads x 16B = 64x32 halves
    int cn = tid >> 2;            // 0..63
    int ck = (tid & 3) << 3;      // 0,8,16,24
    const _Float16* csrc = W + (size_t)(blockN0 + cn) * NIN_ + ck;
    _Float16* cdst[2] = { &ldsB[0][cn * 32 + ck], &ldsB[1][cn * 32 + ck] };

    copy16_g2l(csrc, cdst[0]);                            // prefetch k-step 0
    v8f c00 = {}, c01 = {}, c10 = {}, c11 = {};
    const int KSTEPS = NIN_ / 32;                         // 32
    for (int s = 0; s < KSTEPS; ++s) {
        int k0 = s << 5;
        const _Float16* bufc = ldsB[s & 1];
        if (s + 1 < KSTEPS) {
            copy16_g2l(csrc + (size_t)(s + 1) * 32, cdst[(s + 1) & 1]);
            WAIT_ASYNC(1);                                // oldest copy (current buf) done
        } else {
            WAIT_ASYNC(0);
        }
        __syncthreads();
        v16h a0 = load_a_frag(A, NIN_, m0,      k0, lane);
        v16h a1 = load_a_frag(A, NIN_, m0 + 16, k0, lane);
        v16h b0 = load_b_frag_lds(bufc, nL,      lane);
        v16h b1 = load_b_frag_lds(bufc, nL + 16, lane);
        c00 = wmma_f16(a0, b0, c00);
        c01 = wmma_f16(a0, b1, c01);
        c10 = wmma_f16(a1, b0, c10);
        c11 = wmma_f16(a1, b1, c11);
        __syncthreads();                                   // reads done before buf reuse
    }
    store_proj_tile(c00, m0,      n0,      lane, bias, out16, out32, mode);
    store_proj_tile(c01, m0,      n0 + 16, lane, bias, out16, out32, mode);
    store_proj_tile(c10, m0 + 16, n0,      lane, bias, out16, out32, mode);
    store_proj_tile(c11, m0 + 16, n0 + 16, lane, bias, out16, out32, mode);
}

// ---- Attention scores: S[bh] = Q[bh] @ K[bh]^T (f32 into d_out e-region) ----
// Block tile 128M x 64N (8 waves, 4x2), wave tile 32x32; K=64 (2 steps, no LDS).
__global__ __launch_bounds__(256) void qk_kernel(const _Float16* __restrict__ Q,
                                                 const _Float16* __restrict__ K,
                                                 float* __restrict__ S) {
    int lane = threadIdx.x & 31, w = threadIdx.x >> 5;
    int wm = w >> 1, wn = w & 1;
    int idx = blockIdx.x;                 // 32 blocks per (b,h): 4 M-blocks x 8 N-blocks
    int bx = idx & 7, bm = (idx >> 3) & 3, bh = idx >> 5;
    int m0 = (bm << 7) + (wm << 5);
    int n0 = (bx << 6) + (wn << 5);
    const _Float16* Qb = Q + (size_t)bh * NE_ * HD_;
    const _Float16* Kb = K + (size_t)bh * NE_ * HD_;
    v8f c00 = {}, c01 = {}, c10 = {}, c11 = {};
    #pragma unroll
    for (int k0 = 0; k0 < HD_; k0 += 32) {
        v16h a0 = load_a_frag(Qb, HD_, m0,      k0, lane);
        v16h a1 = load_a_frag(Qb, HD_, m0 + 16, k0, lane);
        v16h b0 = load_b_frag(Kb, HD_, n0,      k0, lane);
        v16h b1 = load_b_frag(Kb, HD_, n0 + 16, k0, lane);
        c00 = wmma_f16(a0, b0, c00);
        c01 = wmma_f16(a0, b1, c01);
        c10 = wmma_f16(a1, b0, c10);
        c11 = wmma_f16(a1, b1, c11);
    }
    float* Sb = S + (size_t)bh * NE_ * NE_;
    int nc = lane & 15, mb = (lane >> 4) << 3;
    #pragma unroll
    for (int r = 0; r < 8; ++r) {
        Sb[(size_t)(m0 + mb + r)      * NE_ + n0 + nc]      = c00[r];
        Sb[(size_t)(m0 + mb + r)      * NE_ + n0 + 16 + nc] = c01[r];
        Sb[(size_t)(m0 + 16 + mb + r) * NE_ + n0 + nc]      = c10[r];
        Sb[(size_t)(m0 + 16 + mb + r) * NE_ + n0 + 16 + nc] = c11[r];
    }
}

// ---- Masked softmax over rows of 512, in place; also emit f16 copy ----------
__global__ __launch_bounds__(256) void softmax_kernel(float* __restrict__ S,
                                                      const int* __restrict__ mask,
                                                      _Float16* __restrict__ E16) {
    int row = blockIdx.x;                 // B*H*NE rows
    int bh = row / NE_;
    int q  = row - bh * NE_;
    int b  = bh >> 4;
    float*     s    = S   + (size_t)row * NE_;
    _Float16*  e16  = E16 + (size_t)row * NE_;
    const int* mrow = mask + ((size_t)b * NE_ + q) * NE_;
    int t = threadIdx.x, lane = t & 31, wave = t >> 5;
    const float NEGINF = -__builtin_inff();
    float x0 = mrow[t]       ? NEGINF : s[t]       * 0.125f;   // 1/sqrt(64)
    float x1 = mrow[t + 256] ? NEGINF : s[t + 256] * 0.125f;

    __shared__ float red[8];
    float m = fmaxf(x0, x1);
    #pragma unroll
    for (int off = 16; off > 0; off >>= 1) m = fmaxf(m, __shfl_xor(m, off, 32));
    if (lane == 0) red[wave] = m;
    __syncthreads();
    if (wave == 0) {
        m = (lane < 8) ? red[lane] : NEGINF;
        #pragma unroll
        for (int off = 4; off > 0; off >>= 1) m = fmaxf(m, __shfl_xor(m, off, 32));
        if (lane == 0) red[0] = m;
    }
    __syncthreads();
    float mx = red[0];
    float p0 = __expf(x0 - mx), p1 = __expf(x1 - mx);
    float sum = p0 + p1;
    __syncthreads();
    #pragma unroll
    for (int off = 16; off > 0; off >>= 1) sum += __shfl_xor(sum, off, 32);
    if (lane == 0) red[wave] = sum;
    __syncthreads();
    if (wave == 0) {
        sum = (lane < 8) ? red[lane] : 0.f;
        #pragma unroll
        for (int off = 4; off > 0; off >>= 1) sum += __shfl_xor(sum, off, 32);
        if (lane == 0) red[0] = sum;
    }
    __syncthreads();
    float inv = 1.f / red[0];
    float e0 = p0 * inv, e1 = p1 * inv;
    s[t] = e0;  s[t + 256] = e1;
    e16[t] = (_Float16)e0;  e16[t + 256] = (_Float16)e1;
}

// ---- out_h = e @ V : [512x512] @ [512x64] per (b,h), concat heads -----------
// Block covers full N=64; block tile 128M x 64N, wave tile 32x32, K=512 staged.
__global__ __launch_bounds__(256) void pv_kernel(const _Float16* __restrict__ E16,
                                                 const _Float16* __restrict__ Vt,
                                                 _Float16* __restrict__ attn) {
    __shared__ _Float16 ldsB[2][64 * 32];
    int tid = threadIdx.x;
    int lane = tid & 31, w = tid >> 5;
    int wm = w >> 1, wn = w & 1;
    int bm = blockIdx.x & 3, bh = blockIdx.x >> 2;       // 4 M-blocks per (b,h)
    int m0 = (bm << 7) + (wm << 5);
    int nL = wn << 5;
    const _Float16* Eb = E16 + (size_t)bh * NE_ * NE_;
    const _Float16* Vb = Vt  + (size_t)bh * HD_ * NE_;

    int cn = tid >> 2;
    int ck = (tid & 3) << 3;
    const _Float16* csrc = Vb + (size_t)cn * NE_ + ck;   // Vt rows = d (64), ld = 512
    _Float16* cdst[2] = { &ldsB[0][cn * 32 + ck], &ldsB[1][cn * 32 + ck] };

    copy16_g2l(csrc, cdst[0]);
    v8f c00 = {}, c01 = {}, c10 = {}, c11 = {};
    const int KSTEPS = NE_ / 32;                          // 16
    for (int s = 0; s < KSTEPS; ++s) {
        int k0 = s << 5;
        const _Float16* bufc = ldsB[s & 1];
        if (s + 1 < KSTEPS) {
            copy16_g2l(csrc + (size_t)(s + 1) * 32, cdst[(s + 1) & 1]);
            WAIT_ASYNC(1);
        } else {
            WAIT_ASYNC(0);
        }
        __syncthreads();
        v16h a0 = load_a_frag(Eb, NE_, m0,      k0, lane);
        v16h a1 = load_a_frag(Eb, NE_, m0 + 16, k0, lane);
        v16h b0 = load_b_frag_lds(bufc, nL,      lane);
        v16h b1 = load_b_frag_lds(bufc, nL + 16, lane);
        c00 = wmma_f16(a0, b0, c00);
        c01 = wmma_f16(a0, b1, c01);
        c10 = wmma_f16(a1, b0, c10);
        c11 = wmma_f16(a1, b1, c11);
        __syncthreads();
    }
    int b = bh >> 4, h = bh & 15;
    int nc = lane & 15, mb = (lane >> 4) << 3;
    #pragma unroll
    for (int r = 0; r < 8; ++r) {
        int mA = m0 + mb + r, mB = m0 + 16 + mb + r;
        attn[((size_t)(b * NE_ + mA)) * (H_ * HD_) + h * HD_ + nL + nc]      = (_Float16)c00[r];
        attn[((size_t)(b * NE_ + mA)) * (H_ * HD_) + h * HD_ + nL + 16 + nc] = (_Float16)c01[r];
        attn[((size_t)(b * NE_ + mB)) * (H_ * HD_) + h * HD_ + nL + nc]      = (_Float16)c10[r];
        attn[((size_t)(b * NE_ + mB)) * (H_ * HD_) + h * HD_ + nL + 16 + nc] = (_Float16)c11[r];
    }
}

// ---- Host launch -------------------------------------------------------------
extern "C" void kernel_launch(void* const* d_in, const int* in_sizes, int n_in,
                              void* d_out, int out_size, void* d_ws, size_t ws_size,
                              hipStream_t stream) {
    const float* events = (const float*)d_in[0];
    const int*   mask   = (const int*)  d_in[1];
    const float* n1a    = (const float*)d_in[2];
    const float* n1b    = (const float*)d_in[3];
    const float* WQw    = (const float*)d_in[4];
    const float* WQb    = (const float*)d_in[5];
    const float* WKw    = (const float*)d_in[6];
    const float* WKb    = (const float*)d_in[7];
    const float* WVw    = (const float*)d_in[8];
    const float* WVb    = (const float*)d_in[9];
    const float* WOw    = (const float*)d_in[10];
    const float* WOb    = (const float*)d_in[11];
    const float* n2a    = (const float*)d_in[12];
    const float* n2b    = (const float*)d_in[13];

    float* out = (float*)d_out;           // [B, NE, NOUT] f32
    float* e32 = out + EOFF_;             // [B, H, NE, NE] f32 (scores, then e, in place)

    // Workspace carve-out (~248 MB)
    char* ws = (char*)d_ws;
    size_t off = 0;
    auto take = [&](size_t bytes) { char* p = ws + off; off += (bytes + 255) & ~(size_t)255; return p; };
    _Float16* x16   = (_Float16*)take((size_t)M_ * NIN_ * 2);
    _Float16* wq16  = (_Float16*)take((size_t)NIN_ * NIN_ * 2);
    _Float16* wk16  = (_Float16*)take((size_t)NIN_ * NIN_ * 2);
    _Float16* wv16  = (_Float16*)take((size_t)NIN_ * NIN_ * 2);
    _Float16* wo16  = (_Float16*)take((size_t)NOUT_ * NIN_ * 2);
    _Float16* Q16   = (_Float16*)take((size_t)B_ * H_ * NE_ * HD_ * 2);
    _Float16* K16   = (_Float16*)take((size_t)B_ * H_ * NE_ * HD_ * 2);
    _Float16* Vt16  = (_Float16*)take((size_t)B_ * H_ * HD_ * NE_ * 2);
    _Float16* att16 = (_Float16*)take((size_t)M_ * H_ * HD_ * 2);
    _Float16* E16   = (_Float16*)take((size_t)B_ * H_ * NE_ * NE_ * 2);
    float*    Y32   = (float*)   take((size_t)M_ * NOUT_ * 4);
    (void)ws_size; (void)n_in; (void)in_sizes; (void)out_size;

    // 1. LayerNorm 1 -> f16 activations
    norm_kernel<_Float16><<<M_, 256, 0, stream>>>(events, n1a, n1b, x16, NIN_);

    // 2. Weights -> f16
    cvt_f32_f16<<<4096, 256, 0, stream>>>(WQw, wq16, NIN_ * NIN_);
    cvt_f32_f16<<<4096, 256, 0, stream>>>(WKw, wk16, NIN_ * NIN_);
    cvt_f32_f16<<<4096, 256, 0, stream>>>(WVw, wv16, NIN_ * NIN_);
    cvt_f32_f16<<<4096, 256, 0, stream>>>(WOw, wo16, NOUT_ * NIN_);

    // 3. Q/K/V projections (1024 blocks each: 64 M-blocks x 16 N-blocks)
    gemm_big_kernel<<<1024, 256, 0, stream>>>(x16, wq16, WQb, Q16,  nullptr, 0);
    gemm_big_kernel<<<1024, 256, 0, stream>>>(x16, wk16, WKb, K16,  nullptr, 0);
    gemm_big_kernel<<<1024, 256, 0, stream>>>(x16, wv16, WVb, Vt16, nullptr, 1);

    // 4. Scores = Q K^T straight into d_out's e-region (8192 blocks)
    qk_kernel<<<8192, 256, 0, stream>>>(Q16, K16, e32);

    // 5. Masked softmax in place + f16 copy of e
    softmax_kernel<<<B_ * H_ * NE_, 256, 0, stream>>>(e32, mask, E16);

    // 6. attn = e @ V  (1024 blocks: 4 M-blocks per (b,h))
    pv_kernel<<<1024, 256, 0, stream>>>(E16, Vt16, att16);

    // 7. Output projection (reuses big GEMM, f32 out)
    gemm_big_kernel<<<1024, 256, 0, stream>>>(att16, wo16, WOb, nullptr, Y32, 2);

    // 8. LayerNorm 2 -> final f32 output
    norm_kernel<float><<<M_, 256, 0, stream>>>(Y32, n2a, n2b, out, NOUT_);
}
